// CONTEXTUAL_AUTOENCODER_49117245997170
// MI455X (gfx1250) — matrix operand order repaired and checked by reading, not compiled, not run on hardware
//
#include <hip/hip_runtime.h>
#include <cstdint>

#define B_SZ    2048
#define IN_SZ   9016
#define ATT_SZ  312
#define WEMB_SZ 512
#define VIEW_SZ 16
#define ADIM_SZ 2048
#define EMB_SZ  2048
#define D1_SZ   4096
#define KM_SZ   2360   // ATT + ADIM
#define KMP_SZ  2368   // padded to multiple of 32

typedef __attribute__((ext_vector_type(16))) __bf16 v16bf;
typedef __attribute__((ext_vector_type(8)))  float  v8f;

union FragU { uint32_t u[8]; uint4 q[2]; v16bf v; };

__device__ __forceinline__ unsigned short f2bf(float f) {
  unsigned int u = __float_as_uint(f);
  u += 0x7FFFu + ((u >> 16) & 1u);   // round-to-nearest-even
  return (unsigned short)(u >> 16);
}
__device__ __forceinline__ float bf2f(unsigned short h) {
  return __uint_as_float(((unsigned int)h) << 16);
}
// LDS aperture: flat address bits [31:0] are the LDS byte offset (ISA 10.2).
__device__ __forceinline__ unsigned ldsOff(const void* p) {
  return (unsigned)(unsigned long long)p;
}

// ---------------------------------------------------------------------------
// Tiled bf16 WMMA GEMM:  C[M,N] = act(A[M,K] * Bt[N,K]^T + bias)
// 128 threads = 4 waves arranged 2(M) x 2(N). Block tile 128x128, K-step 32.
// Each wave: 64x64 -> 4 A-frags x 4 B-frags = 16 v_wmma_f32_16x16x32_bf16
// per K-step (16 b128 LDS reads -> 1.0 loads/WMMA).
// Tiles staged via GLOBAL_LOAD_ASYNC_TO_LDS_B128 with double-buffered LDS.
// ---------------------------------------------------------------------------
#define BM 128
#define BN 128
#define BK 32
#define NT 128         // threads per block
#define LDS_STRIDE 48  // 32 + 16 halfword pad; 96B rows keep b128 16B-aligned

__global__ __launch_bounds__(NT, 1) void gemm_bf16_wmma(
    const unsigned short* __restrict__ A, int lda,    // M x K (row major, bf16 bits)
    const unsigned short* __restrict__ Bt, int ldb,   // N x K (row major = B^T)
    const float* __restrict__ bias,
    void* __restrict__ C, int ldc,
    int M, int N, int K, int relu, int out_bf16)
{
  __shared__ unsigned short lA[2][BM * LDS_STRIDE];
  __shared__ unsigned short lB[2][BN * LDS_STRIDE];

  const int tid  = threadIdx.x;
  const int wave = tid >> 5;     // 0..3
  const int lane = tid & 31;
  const int lrow = lane & 15;
  const int lhi  = lane >> 4;
  const int mw   = wave >> 1;    // 0..1 : 64-row strip
  const int nw   = wave & 1;     // 0..1 : 64-col strip

  const long blockM = (long)blockIdx.y * BM;
  const long blockN = (long)blockIdx.x * BN;

  // tile-copy mapping: each thread owns one 32-elem (64B) row of each tile
  const int tRow = tid;          // 0..127
  const bool bInRange = (blockN + tRow) < N;

  const unsigned short* gA0 = A + (blockM + tRow) * (long)lda;
  const unsigned short* gB0 = Bt + (blockN + tRow) * (long)ldb;

  // Zero OOB B rows once (never async-overwritten afterwards).
  if (!bInRange) {
    const uint4 z = make_uint4(0u, 0u, 0u, 0u);
#pragma unroll
    for (int b = 0; b < 2; ++b) {
      uint4* s = reinterpret_cast<uint4*>(&lB[b][tRow * LDS_STRIDE]);
      s[0] = z; s[1] = z; s[2] = z; s[3] = z;
    }
  }

  // Async stage of one K-tile into buffer `buf` (INST_OFFSET applies to both
  // the global and the LDS address, ISA 10.7/15.18).
  auto stage = [&](int kk, int buf) {
    unsigned la = ldsOff(&lA[buf][tRow * LDS_STRIDE]);
    const unsigned short* ga = gA0 + kk;
    asm volatile("global_load_async_to_lds_b128 %0, %1, off"
                 :: "v"(la), "v"(ga) : "memory");
    asm volatile("global_load_async_to_lds_b128 %0, %1, off offset:16"
                 :: "v"(la), "v"(ga) : "memory");
    asm volatile("global_load_async_to_lds_b128 %0, %1, off offset:32"
                 :: "v"(la), "v"(ga) : "memory");
    asm volatile("global_load_async_to_lds_b128 %0, %1, off offset:48"
                 :: "v"(la), "v"(ga) : "memory");
    if (bInRange) {
      unsigned lb = ldsOff(&lB[buf][tRow * LDS_STRIDE]);
      const unsigned short* gb = gB0 + kk;
      asm volatile("global_load_async_to_lds_b128 %0, %1, off"
                   :: "v"(lb), "v"(gb) : "memory");
      asm volatile("global_load_async_to_lds_b128 %0, %1, off offset:16"
                   :: "v"(lb), "v"(gb) : "memory");
      asm volatile("global_load_async_to_lds_b128 %0, %1, off offset:32"
                   :: "v"(lb), "v"(gb) : "memory");
      asm volatile("global_load_async_to_lds_b128 %0, %1, off offset:48"
                   :: "v"(lb), "v"(gb) : "memory");
    }
  };

  v8f acc[4][4];
#pragma unroll
  for (int t = 0; t < 4; ++t)
#pragma unroll
    for (int j = 0; j < 4; ++j) acc[t][j] = v8f{};

  stage(0, 0);  // prologue
  int buf = 0;

  for (int kk = 0; kk < K; kk += BK) {
    asm volatile("s_wait_asynccnt 0" ::: "memory");
    __syncthreads();  // all waves' tile `kk` resident; prev compute done

    if (kk + BK < K) {
      stage(kk + BK, buf ^ 1);
      __builtin_prefetch(gA0 + kk + 2 * BK, 0, 1);  // warm L2 two tiles ahead
      if (bInRange) __builtin_prefetch(gB0 + kk + 2 * BK, 0, 1);
    }

    // ---- load ALL fragments first so the 16 WMMAs issue back-to-back ----
    // A frag (16x32 bf16): lane(g=lhi,m=lrow): elems0..7 K=g*8.. ; 8..15 K=16+g*8..
    FragU fa[4];
#pragma unroll
    for (int t = 0; t < 4; ++t) {
      const uint4* p = reinterpret_cast<const uint4*>(
          &lA[buf][(mw * 64 + t * 16 + lrow) * LDS_STRIDE + lhi * 8]);
      fa[t].q[0] = p[0];
      fa[t].q[1] = p[2];  // +32 bytes
    }
    // B frag (32x16 bf16): lane n=lrow, elems0..7 K=lhi*8.. ; 8..15 K=16+lhi*8..
    FragU fb[4];
#pragma unroll
    for (int j = 0; j < 4; ++j) {
      const uint4* p = reinterpret_cast<const uint4*>(
          &lB[buf][(nw * 64 + j * 16 + lrow) * LDS_STRIDE + lhi * 8]);
      fb[j].q[0] = p[0];
      fb[j].q[1] = p[2];
    }
#pragma unroll
    for (int t = 0; t < 4; ++t)
#pragma unroll
      for (int j = 0; j < 4; ++j)
        acc[t][j] = __builtin_amdgcn_wmma_f32_16x16x32_bf16(
            false, fa[t].v, false, fb[j].v, (short)0, acc[t][j], false, false);

    buf ^= 1;
  }

  // ---- epilogue: C/D layout: VGPR r, lane -> (m = r + 8*lhi, n = lrow) ----
#pragma unroll
  for (int j = 0; j < 4; ++j) {
    long n = blockN + nw * 64 + j * 16 + lrow;
    if (n < N) {
      float bv = bias ? bias[n] : 0.0f;
#pragma unroll
      for (int t = 0; t < 4; ++t) {
#pragma unroll
        for (int r = 0; r < 8; ++r) {
          long m = blockM + mw * 64 + t * 16 + r + 8 * lhi;
          float val = acc[t][j][r] + bv;
          if (relu) val = fmaxf(val, 0.0f);
          if (out_bf16)
            ((unsigned short*)C)[m * (long)ldc + n] = f2bf(val);
          else
            ((float*)C)[m * (long)ldc + n] = val;
        }
      }
    }
  }
}

// ---------------------------------------------------------------------------
// Weight convert f32 -> bf16 with transpose: Wt[n*Kpad + k] = W[k*N + n], pad 0
// ---------------------------------------------------------------------------
__global__ void transpose_convert(const float* __restrict__ W,
                                  unsigned short* __restrict__ Wt,
                                  int K, int N, int Kpad) {
  const long total = (long)N * Kpad;
  for (long i = blockIdx.x * (long)blockDim.x + threadIdx.x; i < total;
       i += (long)gridDim.x * blockDim.x) {
    long n = i / Kpad;
    long k = i - n * Kpad;
    Wt[i] = (k < K) ? f2bf(W[k * (long)N + n]) : (unsigned short)0;
  }
}

// ---------------------------------------------------------------------------
// Pack activations: desc (B x 512) and gpt ((B*16) x 512) slices of x -> bf16
// ---------------------------------------------------------------------------
__global__ void pack_x(const float* __restrict__ x,
                       unsigned short* __restrict__ desc_bf,
                       unsigned short* __restrict__ gpt_bf) {
  const long total1 = (long)B_SZ * WEMB_SZ;
  const long total2 = (long)B_SZ * VIEW_SZ * WEMB_SZ;
  for (long i = blockIdx.x * (long)blockDim.x + threadIdx.x; i < total1 + total2;
       i += (long)gridDim.x * blockDim.x) {
    if (i < total1) {
      long b = i / WEMB_SZ;
      long c = i - b * WEMB_SZ;
      desc_bf[i] = f2bf(x[b * IN_SZ + ATT_SZ + c]);
    } else {
      long j = i - total1;
      long b = j / (VIEW_SZ * WEMB_SZ);
      long off = j - b * (VIEW_SZ * WEMB_SZ);  // contiguous 16*512 region per row
      gpt_bf[j] = f2bf(x[b * IN_SZ + ATT_SZ + WEMB_SZ + off]);
    }
  }
}

__device__ __forceinline__ float block_reduce_sum(float val, float* red, int t) {
  red[t] = val;
  __syncthreads();
  for (int s = 128; s > 0; s >>= 1) {
    if (t < s) red[t] += red[t + s];
    __syncthreads();
  }
  float r = red[0];
  __syncthreads();
  return r;
}

// ---------------------------------------------------------------------------
// Per-row attention stats: dot(k,q), |q|^2, |k|^2 -> softmax(cs*ed) over 16
// views; accumulate mean attention (div by B folded into atomicAdd).
// ed = sqrt(|q|^2 + |k|^2 - 2 dot)  (exact identity for ||q-k||)
// ---------------------------------------------------------------------------
__global__ __launch_bounds__(256) void attn_stats(
    const float* __restrict__ q, const unsigned short* __restrict__ kbf,
    float* __restrict__ am) {
  __shared__ float red[256];
  __shared__ float s_dot[VIEW_SZ];
  __shared__ float s_kn2[VIEW_SZ];
  __shared__ float s_qn2;
  const int b = blockIdx.x;
  const int t = threadIdx.x;
  const float* qb = q + (long)b * ADIM_SZ;

  float p = 0.f;
  for (int a = t; a < ADIM_SZ; a += 256) { float v = qb[a]; p += v * v; }
  float qn2 = block_reduce_sum(p, red, t);
  if (t == 0) s_qn2 = qn2;

  for (int v = 0; v < VIEW_SZ; ++v) {
    const unsigned short* kv = kbf + ((long)b * VIEW_SZ + v) * ADIM_SZ;
    float pd = 0.f, pk = 0.f;
    for (int a = t; a < ADIM_SZ; a += 256) {
      float kf = bf2f(kv[a]);
      float qf = qb[a];
      pd += kf * qf;
      pk += kf * kf;
    }
    float d  = block_reduce_sum(pd, red, t);
    float k2 = block_reduce_sum(pk, red, t);
    if (t == 0) { s_dot[v] = d; s_kn2[v] = k2; }
  }
  __syncthreads();

  if (t == 0) {
    float qn2v = s_qn2;
    float qn = fmaxf(sqrtf(qn2v), 1e-8f);
    float sc[VIEW_SZ];
    float mx = -3.0e38f;
    for (int v = 0; v < VIEW_SZ; ++v) {
      float kn2 = s_kn2[v];
      float kn = fmaxf(sqrtf(kn2), 1e-8f);
      float cs = s_dot[v] / (qn * kn);
      float ed = sqrtf(fmaxf(qn2v + kn2 - 2.f * s_dot[v], 0.f));
      sc[v] = cs * ed;
      mx = fmaxf(mx, sc[v]);
    }
    float sum = 0.f;
    for (int v = 0; v < VIEW_SZ; ++v) { sc[v] = __expf(sc[v] - mx); sum += sc[v]; }
    float inv = 1.0f / (sum * (float)B_SZ);
    for (int v = 0; v < VIEW_SZ; ++v) atomicAdd(&am[v], sc[v] * inv);
  }
}

__global__ void init_am(float* am) {
  if (threadIdx.x < VIEW_SZ) am[threadIdx.x] = 0.f;
}

// ---------------------------------------------------------------------------
// Build concat input for Wm GEMM: [att_emb(312) | fused(2048) | pad(8)] bf16
// fused[b,a] = sum_v am[v] * v[b,v,a]
// ---------------------------------------------------------------------------
__global__ __launch_bounds__(256) void fused_concat(
    const float* __restrict__ x, const unsigned short* __restrict__ vbf,
    const float* __restrict__ am, unsigned short* __restrict__ cc) {
  __shared__ float s_am[VIEW_SZ];
  const int b = blockIdx.x;
  const int t = threadIdx.x;
  if (t < VIEW_SZ) s_am[t] = am[t];
  __syncthreads();
  const float* xb = x + (long)b * IN_SZ;
  unsigned short* cb = cc + (long)b * KMP_SZ;
  for (int c = t; c < ATT_SZ; c += 256) cb[c] = f2bf(xb[c]);
  const unsigned short* vb = vbf + (long)b * VIEW_SZ * ADIM_SZ;
  for (int a = t; a < ADIM_SZ; a += 256) {
    float s = 0.f;
#pragma unroll
    for (int v = 0; v < VIEW_SZ; ++v) s += s_am[v] * bf2f(vb[(long)v * ADIM_SZ + a]);
    cb[ATT_SZ + a] = f2bf(s);
  }
  for (int c = KM_SZ + t; c < KMP_SZ; c += 256) cb[c] = 0;
}

// ---------------------------------------------------------------------------
extern "C" void kernel_launch(void* const* d_in, const int* in_sizes, int n_in,
                              void* d_out, int out_size, void* d_ws, size_t ws_size,
                              hipStream_t stream) {
  (void)in_sizes; (void)n_in; (void)out_size; (void)ws_size;
  const float* x   = (const float*)d_in[0];
  const float* Wq  = (const float*)d_in[1];
  const float* bq  = (const float*)d_in[2];
  const float* Wk  = (const float*)d_in[3];
  const float* bk  = (const float*)d_in[4];
  const float* Wv  = (const float*)d_in[5];
  const float* bv  = (const float*)d_in[6];
  const float* Wm  = (const float*)d_in[7];
  const float* bm  = (const float*)d_in[8];
  const float* Wd1 = (const float*)d_in[9];
  const float* bd1 = (const float*)d_in[10];
  const float* Wd2 = (const float*)d_in[11];
  const float* bd2 = (const float*)d_in[12];

  char* p = (char*)d_ws;
  auto carve = [&](size_t bytes) -> void* {
    void* r = (void*)p;
    p += (bytes + 255) & ~(size_t)255;
    return r;
  };
  unsigned short* wq_t  = (unsigned short*)carve((size_t)ADIM_SZ * WEMB_SZ * 2);
  unsigned short* wk_t  = (unsigned short*)carve((size_t)ADIM_SZ * WEMB_SZ * 2);
  unsigned short* wv_t  = (unsigned short*)carve((size_t)ADIM_SZ * WEMB_SZ * 2);
  unsigned short* wm_t  = (unsigned short*)carve((size_t)EMB_SZ * KMP_SZ * 2);
  unsigned short* wd1_t = (unsigned short*)carve((size_t)D1_SZ * EMB_SZ * 2);
  unsigned short* wd2_t = (unsigned short*)carve((size_t)IN_SZ * D1_SZ * 2);
  unsigned short* desc  = (unsigned short*)carve((size_t)B_SZ * WEMB_SZ * 2);
  unsigned short* gpt   = (unsigned short*)carve((size_t)B_SZ * VIEW_SZ * WEMB_SZ * 2);
  float*          qf    = (float*)carve((size_t)B_SZ * ADIM_SZ * 4);
  unsigned short* kbf   = (unsigned short*)carve((size_t)B_SZ * VIEW_SZ * ADIM_SZ * 2);
  unsigned short* vbf   = (unsigned short*)carve((size_t)B_SZ * VIEW_SZ * ADIM_SZ * 2);
  float*          am    = (float*)carve((size_t)VIEW_SZ * 4);
  unsigned short* cc    = (unsigned short*)carve((size_t)B_SZ * KMP_SZ * 2);
  unsigned short* zbf   = (unsigned short*)carve((size_t)B_SZ * EMB_SZ * 2);
  unsigned short* hbf   = (unsigned short*)carve((size_t)B_SZ * D1_SZ * 2);

  // one-time conversions (cheap relative to GEMMs)
  transpose_convert<<<1024, 256, 0, stream>>>(Wq, wq_t, WEMB_SZ, ADIM_SZ, WEMB_SZ);
  transpose_convert<<<1024, 256, 0, stream>>>(Wk, wk_t, WEMB_SZ, ADIM_SZ, WEMB_SZ);
  transpose_convert<<<1024, 256, 0, stream>>>(Wv, wv_t, WEMB_SZ, ADIM_SZ, WEMB_SZ);
  transpose_convert<<<2048, 256, 0, stream>>>(Wm, wm_t, KM_SZ, EMB_SZ, KMP_SZ);
  transpose_convert<<<2048, 256, 0, stream>>>(Wd1, wd1_t, EMB_SZ, D1_SZ, EMB_SZ);
  transpose_convert<<<4096, 256, 0, stream>>>(Wd2, wd2_t, D1_SZ, IN_SZ, D1_SZ);
  pack_x<<<2048, 256, 0, stream>>>(x, desc, gpt);

  dim3 blk(NT);
  // q = desc @ Wq + bq  (f32 out; feeds norm/dot reductions)
  gemm_bf16_wmma<<<dim3(ADIM_SZ / BN, B_SZ / BM), blk, 0, stream>>>(
      desc, WEMB_SZ, wq_t, WEMB_SZ, bq, qf, ADIM_SZ,
      B_SZ, ADIM_SZ, WEMB_SZ, 0, 0);
  // k = gpt @ Wk + bk  (bf16 out)
  gemm_bf16_wmma<<<dim3(ADIM_SZ / BN, (B_SZ * VIEW_SZ) / BM), blk, 0, stream>>>(
      gpt, WEMB_SZ, wk_t, WEMB_SZ, bk, kbf, ADIM_SZ,
      B_SZ * VIEW_SZ, ADIM_SZ, WEMB_SZ, 0, 1);
  // v = gpt @ Wv + bv  (bf16 out)
  gemm_bf16_wmma<<<dim3(ADIM_SZ / BN, (B_SZ * VIEW_SZ) / BM), blk, 0, stream>>>(
      gpt, WEMB_SZ, wv_t, WEMB_SZ, bv, vbf, ADIM_SZ,
      B_SZ * VIEW_SZ, ADIM_SZ, WEMB_SZ, 0, 1);

  init_am<<<1, 32, 0, stream>>>(am);
  attn_stats<<<B_SZ, 256, 0, stream>>>(qf, kbf, am);
  fused_concat<<<B_SZ, 256, 0, stream>>>(x, vbf, am, cc);

  // z = relu(cc @ Wm + bm)  (bf16 out)
  gemm_bf16_wmma<<<dim3(EMB_SZ / BN, B_SZ / BM), blk, 0, stream>>>(
      cc, KMP_SZ, wm_t, KMP_SZ, bm, zbf, EMB_SZ,
      B_SZ, EMB_SZ, KMP_SZ, 1, 1);
  // h = relu(z @ Wd1 + bd1)  (bf16 out)
  gemm_bf16_wmma<<<dim3(D1_SZ / BN, B_SZ / BM), blk, 0, stream>>>(
      zbf, EMB_SZ, wd1_t, EMB_SZ, bd1, hbf, D1_SZ,
      B_SZ, D1_SZ, EMB_SZ, 1, 1);
  // out = h @ Wd2 + bd2  (f32 -> d_out; N=9016 handled by bounds checks)
  gemm_bf16_wmma<<<dim3((IN_SZ + BN - 1) / BN, B_SZ / BM), blk, 0, stream>>>(
      hbf, D1_SZ, wd2_t, D1_SZ, bd2, d_out, IN_SZ,
      B_SZ, IN_SZ, D1_SZ, 0, 0);
}
